// ModelLocFrame_69844758168153
// MI455X (gfx1250) — matrix operand order repaired and checked by reading, not compiled, not run on hardware
//
#include <hip/hip_runtime.h>

// ---- problem constants ----
#define BB      4
#define NN      16384
#define NTYPES  2
#define NNEI_A  80
#define NNEI_R  16
#define NNEI    96
#define NDESC   336
#define NDESC_P 352      // padded K for layer0 (11 * 32)
#define KT0     11       // k-tiles layer0
#define KT12    8        // k-tiles layers 1/2 (240 -> 256)
#define HH      240
#define NT      16       // n-tiles (H padded to 256)
#define RMIN_   0.1f
#define EPS_    1e-6f
#define NAT     8192     // NN / NTYPES
#define ROWS_T  32768    // BB * NAT rows per type
#define DSTR    360      // LDS stride (halves) for desc tile: bank-conflict free
#define HSTR    264      // LDS stride (halves) for h tile: bank-conflict free

typedef __attribute__((ext_vector_type(16))) _Float16 v16h;
typedef __attribute__((ext_vector_type(8)))  float    v8f;

union Frag16 { uint4 u[2]; v16h v; };
struct __align__(8) H4 { _Float16 a, b, c, d; };

// ---------------------------------------------------------------------------
// Repack weights (K x 240, row-major f32) into B fragments in this code's contiguous k order (as load_a):
//   dst[t][kt][nt][lane][e] ; k = 32*kt + 16*(lane/16) + e ; n = 16*nt + lane%16
// K padded with zeros, N padded 240 -> 256 with zeros.
// ---------------------------------------------------------------------------
__global__ void packB_kernel(const float* __restrict__ W, _Float16* __restrict__ dst,
                             int K, int KT, int total) {
    int id = blockIdx.x * 256 + threadIdx.x;
    if (id >= total) return;
    int e    = id & 15;
    int lane = (id >> 4) & 31;
    int nt   = (id >> 9) & 15;
    int ktt  = id >> 13;           // t*KT + kt
    int kt   = ktt % KT;
    int t    = ktt / KT;
    int k    = kt * 32 + ((lane >> 4) << 4) + e;
    int n    = nt * 16 + (lane & 15);
    float v = 0.0f;
    if (k < K && n < HH) v = W[((size_t)t * K + k) * HH + n];
    dst[id] = (_Float16)v;
}

// ---------------------------------------------------------------------------
// Descriptor stage: one wave32 per atom. Lane l handles neighbors l, l+32, l+64
// (lanes 16..31 of the 3rd pass handle the 16 radial neighbors).
// Output: normalized descriptor rows, f16, padded to 352, row = t*32768 + b*8192 + i
// ---------------------------------------------------------------------------
__global__ void __launch_bounds__(256) desc_kernel(
        const float* __restrict__ coord, const int* __restrict__ nlist,
        const float* __restrict__ avg,   const float* __restrict__ stdv,
        _Float16* __restrict__ descws) {
    int atom = blockIdx.x * 8 + (threadIdx.x >> 5);
    int lane = threadIdx.x & 31;
    int b = atom >> 14;          // / 16384
    int n = atom & 16383;
    const float* cb = coord + (size_t)b * NN * 3;
    float cx = cb[n * 3 + 0], cy = cb[n * 3 + 1], cz = cb[n * 3 + 2];
    const int* nl = nlist + (size_t)atom * NNEI;

    // local frame from neighbors 0 and 1 (redundant per lane; L1/L2 broadcast)
    int i0 = nl[0], i1 = nl[1];
    float r0x = cb[i0*3+0]-cx, r0y = cb[i0*3+1]-cy, r0z = cb[i0*3+2]-cz;
    float r1x = cb[i1*3+0]-cx, r1y = cb[i1*3+1]-cy, r1z = cb[i1*3+2]-cz;
    float n0 = sqrtf(r0x*r0x + r0y*r0y + r0z*r0z);
    float s0 = 1.0f / (n0 + EPS_);
    float e1x = r0x*s0, e1y = r0y*s0, e1z = r0z*s0;
    float d1 = r1x*e1x + r1y*e1y + r1z*e1z;
    float ux = r1x - d1*e1x, uy = r1y - d1*e1y, uz = r1z - d1*e1z;
    float nu = sqrtf(ux*ux + uy*uy + uz*uz);
    float su = 1.0f / (nu + EPS_);
    float e2x = ux*su, e2y = uy*su, e2z = uz*su;
    float e3x = e1y*e2z - e1z*e2y;
    float e3y = e1z*e2x - e1x*e2z;
    float e3z = e1x*e2y - e1y*e2x;

    int t   = n >> 13;                  // type (atoms type-sorted within row)
    int row = b * NAT + (n & 8191);     // row within type
    size_t base = (size_t)(t * ROWS_T + row) * NDESC_P;
    const float* av = avg  + t * NDESC;
    const float* sd = stdv + t * NDESC;

    #pragma unroll
    for (int it = 0; it < 3; ++it) {
        int j = lane + 32 * it;
        int idx = nl[j];
        float dx = cb[idx*3+0]-cx, dy = cb[idx*3+1]-cy, dz = cb[idx*3+2]-cz;
        float rr = sqrtf(dx*dx + dy*dy + dz*dz);
        float ra = fmaxf(rr, RMIN_);
        if (j < NNEI_A) {
            float inv  = 1.0f / ra;
            float inv2 = inv * inv;
            float v0 = inv;
            float v1 = (e1x*dx + e1y*dy + e1z*dz) * inv2;
            float v2 = (e2x*dx + e2y*dy + e2z*dz) * inv2;
            float v3 = (e3x*dx + e3y*dy + e3z*dz) * inv2;
            int k = 4 * j;
            H4 h;
            h.a = (_Float16)((v0 - av[k+0]) / sd[k+0]);
            h.b = (_Float16)((v1 - av[k+1]) / sd[k+1]);
            h.c = (_Float16)((v2 - av[k+2]) / sd[k+2]);
            h.d = (_Float16)((v3 - av[k+3]) / sd[k+3]);
            *(H4*)(descws + base + k) = h;     // coalesced 8B stores
        } else {
            int jr = j - NNEI_A;               // radial 0..15 (it==2, lanes 16..31)
            int k  = 4 * NNEI_A + jr;
            float v = (1.0f / ra - av[k]) / sd[k];
            descws[base + k] = (_Float16)v;
        }
        if (it == 2 && lane < 16)              // zero K-padding 336..351
            descws[base + NDESC + lane] = (_Float16)0.0f;
    }
}

// ---------------------------------------------------------------------------
// Fused 3-layer MLP + head. 256 threads = 8 waves; 64 rows per workgroup.
// Wave w: row-block rb = w/2 (16 rows), n-tile half = w%2 (8 of 16 n-tiles).
// A-fragments built from LDS tiles; residual kept in C-fragment registers.
// ---------------------------------------------------------------------------
__device__ __forceinline__ v16h load_a(const _Float16* buf, int stride,
                                       int row0, int kt, int g, int m) {
    // A frag: lane (g,m): k = 32*kt + 16*g + e
    const uint4* p = (const uint4*)(buf + (row0 + m) * stride + kt * 32 + g * 16);
    Frag16 f; f.u[0] = p[0]; f.u[1] = p[1];
    return f.v;
}
__device__ __forceinline__ v16h load_b(const _Float16* w, int lane) {
    const uint4* p = (const uint4*)(w + lane * 16);   // 32B contiguous per lane
    Frag16 f; f.u[0] = p[0]; f.u[1] = p[1];
    return f.v;
}

__global__ void __launch_bounds__(256) mlp_kernel(
        const _Float16* __restrict__ descws,
        const _Float16* __restrict__ W0p, const _Float16* __restrict__ W1p,
        const _Float16* __restrict__ W2p,
        const float* __restrict__ b0, const float* __restrict__ b1,
        const float* __restrict__ idt1, const float* __restrict__ b2,
        const float* __restrict__ idt2, const float* __restrict__ Wf,
        const float* __restrict__ bf, float* __restrict__ out) {
    __shared__ __align__(16) _Float16 smem[64 * DSTR];   // 46080B; reused as 64 x HSTR h-tile

    int tid  = threadIdx.x;
    int lane = tid & 31;
    int w    = tid >> 5;
    int rb   = w >> 1;            // row-block 0..3
    int ntS  = (w & 1) * 8;       // n-tile start (8 tiles per wave)
    int t    = blockIdx.y;
    int rowBase = blockIdx.x * 64;
    int g = lane >> 4, m = lane & 15, nloc = lane & 15;

    // stage descriptor tile: 64 rows x 352 halves (44 uint4 chunks / row)
    {
        const uint4* src = (const uint4*)(descws + (size_t)(t * ROWS_T + rowBase) * NDESC_P);
        for (int c = tid; c < 64 * 44; c += 256) {
            int r = c / 44, ch = c % 44;
            uint4 v = src[(size_t)r * 44 + ch];
            *(uint4*)(smem + r * DSTR + ch * 8) = v;
        }
    }
    __syncthreads();

    v8f zf = {};
    v8f accA[8];
    #pragma unroll
    for (int i = 0; i < 8; ++i) accA[i] = zf;

    // ---- layer 0: X(16x352) @ W0(352x256) ----
    const _Float16* W0t = W0p + (size_t)t * (KT0 * NT * 512);
    for (int kt = 0; kt < KT0; ++kt) {
        v16h a = load_a(smem, DSTR, rb * 16, kt, g, m);
        const _Float16* wb = W0t + (size_t)kt * (NT * 512);
        #pragma unroll
        for (int ai = 0; ai < 8; ++ai) {
            v16h bv = load_b(wb + (ntS + ai) * 512, lane);
            accA[ai] = __builtin_amdgcn_wmma_f32_16x16x32_f16(
                false, a, false, bv, (short)0, accA[ai], false, false);
        }
    }
    #pragma unroll
    for (int ai = 0; ai < 8; ++ai) {
        int nn = (ntS + ai) * 16 + nloc;
        float bias = (nn < HH) ? b0[t * HH + nn] : 0.0f;
        #pragma unroll
        for (int r = 0; r < 8; ++r) accA[ai][r] = tanhf(accA[ai][r] + bias);
    }
    __syncthreads();                      // desc tile dead -> reuse LDS as h tile
    #pragma unroll
    for (int ai = 0; ai < 8; ++ai) {
        int nn = (ntS + ai) * 16 + nloc;
        #pragma unroll
        for (int r = 0; r < 8; ++r)
            smem[(rb * 16 + r + 8 * g) * HSTR + nn] = (_Float16)accA[ai][r];
    }
    __syncthreads();

    // ---- layers 1 & 2: h = tanh(h @ W + b) * idt + h ----
    const _Float16* Wlp[2] = { W1p, W2p };
    const float*    blp[2] = { b1, b2 };
    const float*    ilp[2] = { idt1, idt2 };
    #pragma unroll
    for (int layer = 0; layer < 2; ++layer) {
        v8f acc2[8];
        #pragma unroll
        for (int i = 0; i < 8; ++i) acc2[i] = zf;
        const _Float16* Wt = Wlp[layer] + (size_t)t * (KT12 * NT * 512);
        for (int kt = 0; kt < KT12; ++kt) {
            v16h a = load_a(smem, HSTR, rb * 16, kt, g, m);
            const _Float16* wb = Wt + (size_t)kt * (NT * 512);
            #pragma unroll
            for (int ai = 0; ai < 8; ++ai) {
                v16h bv = load_b(wb + (ntS + ai) * 512, lane);
                acc2[ai] = __builtin_amdgcn_wmma_f32_16x16x32_f16(
                    false, a, false, bv, (short)0, acc2[ai], false, false);
            }
        }
        #pragma unroll
        for (int ai = 0; ai < 8; ++ai) {
            int nn = (ntS + ai) * 16 + nloc;
            float bias = (nn < HH) ? blp[layer][t * HH + nn] : 0.0f;
            float idt  = (nn < HH) ? ilp[layer][t * HH + nn] : 0.0f;
            #pragma unroll
            for (int r = 0; r < 8; ++r)
                accA[ai][r] = tanhf(acc2[ai][r] + bias) * idt + accA[ai][r];
        }
        __syncthreads();                  // everyone done reading previous h
        #pragma unroll
        for (int ai = 0; ai < 8; ++ai) {
            int nn = (ntS + ai) * 16 + nloc;
            #pragma unroll
            for (int r = 0; r < 8; ++r)
                smem[(rb * 16 + r + 8 * g) * HSTR + nn] = (_Float16)accA[ai][r];
        }
        __syncthreads();
    }

    // ---- head: energy = h @ Wf + bf ----
    if (tid < 64) {
        int rg = rowBase + tid;           // row within type
        const float* wf = Wf + t * HH;
        float s = bf[t];
        for (int k = 0; k < HH; ++k)
            s += (float)smem[tid * HSTR + k] * wf[k];
        int bo = rg / NAT, io = rg % NAT;
        out[(size_t)bo * NN + t * NAT + io] = s;
    }
}

// ---------------------------------------------------------------------------
extern "C" void kernel_launch(void* const* d_in, const int* in_sizes, int n_in,
                              void* d_out, int out_size, void* d_ws, size_t ws_size,
                              hipStream_t stream) {
    (void)in_sizes; (void)n_in; (void)out_size; (void)ws_size;
    const float* coord = (const float*)d_in[0];
    const float* avg   = (const float*)d_in[1];
    const float* stdv  = (const float*)d_in[2];
    const float* W0    = (const float*)d_in[3];
    const float* b0    = (const float*)d_in[4];
    const float* W1    = (const float*)d_in[5];
    const float* b1    = (const float*)d_in[6];
    const float* idt1  = (const float*)d_in[7];
    const float* W2    = (const float*)d_in[8];
    const float* b2    = (const float*)d_in[9];
    const float* idt2  = (const float*)d_in[10];
    const float* Wf    = (const float*)d_in[11];
    const float* bf    = (const float*)d_in[12];
    const int*   nlist = (const int*)d_in[13];
    float* out = (float*)d_out;

    char* ws = (char*)d_ws;
    _Float16* descws = (_Float16*)ws;
    size_t descBytes = (size_t)NTYPES * ROWS_T * NDESC_P * 2;   // ~46.1 MB
    _Float16* W0p = (_Float16*)(ws + descBytes);
    size_t w0Bytes  = (size_t)NTYPES * KT0  * NT * 512 * 2;
    _Float16* W1p = (_Float16*)(ws + descBytes + w0Bytes);
    size_t w12Bytes = (size_t)NTYPES * KT12 * NT * 512 * 2;
    _Float16* W2p = (_Float16*)(ws + descBytes + w0Bytes + w12Bytes);

    int tot0  = NTYPES * KT0  * NT * 512;
    int tot12 = NTYPES * KT12 * NT * 512;
    packB_kernel<<<(tot0  + 255) / 256, 256, 0, stream>>>(W0, W0p, NDESC, KT0,  tot0);
    packB_kernel<<<(tot12 + 255) / 256, 256, 0, stream>>>(W1, W1p, HH,    KT12, tot12);
    packB_kernel<<<(tot12 + 255) / 256, 256, 0, stream>>>(W2, W2p, HH,    KT12, tot12);

    desc_kernel<<<(BB * NN) / 8, 256, 0, stream>>>(coord, nlist, avg, stdv, descws);

    dim3 grid(ROWS_T / 64, NTYPES);
    mlp_kernel<<<grid, 256, 0, stream>>>(descws, W0p, W1p, W2p,
                                         b0, b1, idt1, b2, idt2, Wf, bf, out);
}
